// AthenaSA_24739011625812
// MI455X (gfx1250) — matrix operand order repaired and checked by reading, not compiled, not run on hardware
//
#include <hip/hip_runtime.h>
#include <hip/hip_bf16.h>
#include <cstdint>

// ---------------- problem constants ----------------
#define BB   2
#define SS   4096
#define EE   2048
#define HH   16
#define HKV  4
#define DKK  128
#define DVV  128
#define WINW 512
#define MT   (BB*SS)          // 8192 tokens

typedef float  v8f   __attribute__((ext_vector_type(8)));
typedef __bf16 bfx8  __attribute__((ext_vector_type(8)));
typedef __bf16 bfx16 __attribute__((ext_vector_type(16)));
typedef int    v4i   __attribute__((ext_vector_type(4)));
typedef __attribute__((address_space(1))) v4i gv4i;   // global 16B vector
typedef __attribute__((address_space(3))) v4i lv4i;   // LDS   16B vector

#if defined(__HIP_DEVICE_COMPILE__) && __has_builtin(__builtin_amdgcn_global_load_async_to_lds_b128)
#define ASYNC_LDS 1
#else
#define ASYNC_LDS 0
#endif

__device__ __forceinline__ v8f wmma_bf16(bfx16 a, bfx16 b, v8f c) {
  return __builtin_amdgcn_wmma_f32_16x16x32_bf16(
      /*neg_a=*/false, a, /*neg_b=*/false, b,
      /*c_mod=*/(short)0, c, /*reuse_a=*/false, /*reuse_b=*/false);
}

__device__ __forceinline__ bfx16 cat8(bfx8 lo, bfx8 hi) {
  return __builtin_shufflevector(lo, hi, 0,1,2,3,4,5,6,7,8,9,10,11,12,13,14,15);
}

// ---------------- 1) RMSNorm -> bf16 ----------------
__global__ __launch_bounds__(256)
void rmsnorm_kernel(const float* __restrict__ emb, __bf16* __restrict__ outp) {
  const int t = blockIdx.x;                     // token
  const float* x = emb + (size_t)t * EE;
  float ss = 0.f;
  for (int i = threadIdx.x; i < EE; i += 256) { float v = x[i]; ss += v * v; }
  #pragma unroll
  for (int o = 16; o > 0; o >>= 1) ss += __shfl_xor(ss, o, 32);
  __shared__ float red[8];
  const int wid = threadIdx.x >> 5, lane = threadIdx.x & 31;
  if (lane == 0) red[wid] = ss;
  __syncthreads();
  if (threadIdx.x == 0) {
    float tot = 0.f;
    #pragma unroll
    for (int i = 0; i < 8; ++i) tot += red[i];
    red[0] = rsqrtf(tot / (float)EE + 1e-5f);
  }
  __syncthreads();
  const float rinv = red[0];
  __bf16* o = outp + (size_t)t * EE;
  for (int i = threadIdx.x; i < EE; i += 256) o[i] = (__bf16)(x[i] * rinv);
}

// ---------------- 2) cast + transpose weights: W[K,N] fp32 -> Wt[N,K] bf16 -----
__global__ __launch_bounds__(256)
void cast_transpose_kernel(const float* __restrict__ W, __bf16* __restrict__ Wt,
                           int K, int N) {
  size_t idx = (size_t)blockIdx.x * 256 + threadIdx.x;
  if (idx >= (size_t)K * N) return;
  int n = (int)(idx % N);
  int k = (int)(idx / N);
  Wt[(size_t)n * K + k] = (__bf16)W[idx];
}

// ---------------- 3) bf16 WMMA GEMM: C[M,N] = A[M,K] * Bt[N,K]^T ---------------
// 128 threads = 4 waves; block tile 64(M) x 128(N); wave computes 16x128.
// B tile (128 n-rows x 32 k) double-buffered in LDS via async-load-to-LDS.
__global__ __launch_bounds__(128)
void gemm_bf16_wmma(const __bf16* __restrict__ A, const __bf16* __restrict__ Bt,
                    float* __restrict__ C, const float* __restrict__ resid,
                    int N, int K) {
  __shared__ __align__(128) __bf16 Bsh[2][128 * 32];   // 2 x 8KB
  const int tid  = threadIdx.x;
  const int lane = tid & 31;
  const int wave = tid >> 5;
  const int l15  = lane & 15;
  const int kh   = lane >> 4;
  const int m0 = blockIdx.y * 64 + wave * 16;
  const int n0 = blockIdx.x * 128;

  // Each thread stages one 64-byte B row (n = n0 + tid) per K-chunk.
  auto stage = [&](int buf, int kc) {
    const __bf16* g = Bt + (size_t)(n0 + tid) * K + kc;
    __bf16* l = &Bsh[buf][tid * 32];
#if ASYNC_LDS
    #pragma unroll
    for (int j = 0; j < 4; ++j)
      __builtin_amdgcn_global_load_async_to_lds_b128(
          (gv4i*)(g + j * 8), (lv4i*)(l + j * 8), 0, 0);
#else
    bfx16 t0 = *(const bfx16*)g;
    bfx16 t1 = *(const bfx16*)(g + 16);
    *(bfx16*)l = t0;
    *(bfx16*)(l + 16) = t1;
#endif
  };

  v8f acc[8] = {};
  stage(0, 0);
#if ASYNC_LDS
  asm volatile("s_wait_asynccnt 0x0" ::: "memory");
#endif
  __syncthreads();

  const __bf16* arow = A + (size_t)(m0 + l15) * K;

  for (int kc = 0; kc < K; kc += 32) {
    const int cur = (kc >> 5) & 1;
    if (kc + 32 < K) stage(cur ^ 1, kc + 32);      // async prefetch next tile

    bfx8 alo = *(const bfx8*)(arow + kc + kh * 16);
    bfx8 ahi = *(const bfx8*)(arow + kc + kh * 16 + 8);
    bfx16 aop = cat8(alo, ahi);
    __builtin_prefetch(arow + kc + 256, 0, 3);     // global_prefetch_b8 for A

    bfx16 bop[8];
    #pragma unroll
    for (int t = 0; t < 8; ++t)
      bop[t] = *(const bfx16*)(&Bsh[cur][(t * 16 + l15) * 32 + kh * 16]);

    #pragma unroll
    for (int t = 0; t < 8; ++t)                    // back-to-back WMMA issue
      acc[t] = wmma_bf16(aop, bop[t], acc[t]);

#if ASYNC_LDS
    asm volatile("s_wait_asynccnt 0x0" ::: "memory");
#endif
    __syncthreads();
  }

  #pragma unroll
  for (int t = 0; t < 8; ++t) {
    #pragma unroll
    for (int r = 0; r < 8; ++r) {
      const int m = m0 + r + 8 * kh;               // C/D layout: M = r + 8*(lane>>4)
      const int n = n0 + t * 16 + l15;             // N = lane&15
      const size_t idx = (size_t)m * N + n;
      float v = acc[t][r];
      if (resid) v += resid[idx];
      C[idx] = v;
    }
  }
}

// ---------------- 4) RoPE + pack: src[B,S,h*DK] fp32 -> dst[B,h,S,DK] bf16 -----
__global__ __launch_bounds__(256)
void rope_pack_kernel(const float* __restrict__ src, const float* __restrict__ cosb,
                      const float* __restrict__ sinb, __bf16* __restrict__ dst,
                      int heads) {
  size_t idx = (size_t)blockIdx.x * 256 + threadIdx.x;
  const size_t total = (size_t)BB * SS * heads * DKK;
  if (idx >= total) return;
  const int d = (int)(idx & (DKK - 1));
  size_t r = idx >> 7;
  const int hh = (int)(r % heads); r /= heads;
  const int s  = (int)(r % SS);
  const int b  = (int)(r / SS);
  const size_t base = ((size_t)(b * SS + s) * heads + hh) * DKK;
  const float f  = src[base + d];
  const float fs = src[base + ((d + DKK / 2) & (DKK - 1))];
  const float c  = cosb[(size_t)s * DKK + d];
  const float sn = sinb[(size_t)s * DKK + d];
  dst[(((size_t)b * heads + hh) * SS + s) * DKK + d] = (__bf16)(c * f + sn * fs);
}

// V: [B,S,HKV*DV] fp32 -> Vt[B,HKV,DV,S] bf16 (transposed for PV B-operand)
__global__ __launch_bounds__(256)
void pack_vT_kernel(const float* __restrict__ v, __bf16* __restrict__ vt) {
  size_t idx = (size_t)blockIdx.x * 256 + threadIdx.x;
  const size_t total = (size_t)BB * SS * HKV * DVV;
  if (idx >= total) return;
  const int d = (int)(idx & (DVV - 1));
  size_t r = idx >> 7;
  const int hh = (int)(r % HKV); r /= HKV;
  const int s  = (int)(r % SS);
  const int b  = (int)(r / SS);
  vt[(((size_t)b * HKV + hh) * DVV + d) * SS + s] = (__bf16)v[idx];
}

// ---------------- 5) sliding-window flash attention ----------------
// grid (S/64, H, B), 128 threads = 4 waves; wave owns one 16-query tile.
__global__ __launch_bounds__(128)
void attn_kernel(const __bf16* __restrict__ Q, const __bf16* __restrict__ Kk,
                 const __bf16* __restrict__ Vt, __bf16* __restrict__ Out) {
  __shared__ __align__(128) __bf16 Plds[4][16 * 32];
  const int lane = threadIdx.x & 31;
  const int wave = threadIdx.x >> 5;
  const int l15  = lane & 15;
  const int kh   = lane >> 4;
  const int q0 = (blockIdx.x * 4 + wave) * 16;
  const int h  = blockIdx.y, b = blockIdx.z;
  const int kvh = h >> 2;                          // H/HKV = 4
  const __bf16* Qp = Q  + ((size_t)(b * HH + h) * SS) * DKK;
  const __bf16* Kp = Kk + ((size_t)(b * HKV + kvh) * SS) * DKK;
  const __bf16* Vp = Vt + ((size_t)(b * HKV + kvh) * DVV) * SS;
  __bf16* Pl = Plds[wave];

  // Q tile in A-operand layout (4 chunks of K=32)
  bfx16 qa[4];
  {
    const __bf16* qrow = Qp + (size_t)(q0 + l15) * DKK;
    #pragma unroll
    for (int c = 0; c < 4; ++c) {
      bfx8 lo = *(const bfx8*)(qrow + c * 32 + kh * 16);
      bfx8 hi = *(const bfx8*)(qrow + c * 32 + kh * 16 + 8);
      qa[c] = cat8(lo, hi);
    }
  }

  v8f Oc[8] = {};
  float mrow[8], lrow[8];
  #pragma unroll
  for (int r = 0; r < 8; ++r) { mrow[r] = -INFINITY; lrow[r] = 0.f; }

  int ks = q0 - WINW + 1; if (ks < 0) ks = 0; ks &= ~31;
  const float scl = 0.08838834764831845f;          // 1/sqrt(128)

  for (int jt = ks; jt <= q0 + 15; jt += 32) {
    // ---- load all K operands first, then 8 back-to-back WMMAs ----
    bfx16 kb0[4], kb1[4];
    #pragma unroll
    for (int c = 0; c < 4; ++c) {
      kb0[c] = *(const bfx16*)(Kp + (size_t)(jt + l15) * DKK + c * 32 + kh * 16);
      kb1[c] = *(const bfx16*)(Kp + (size_t)(jt + 16 + l15) * DKK + c * 32 + kh * 16);
    }
    v8f s0 = {}, s1 = {};
    #pragma unroll
    for (int c = 0; c < 4; ++c) {
      s0 = wmma_bf16(qa[c], kb0[c], s0);
      s1 = wmma_bf16(qa[c], kb1[c], s1);
    }
    // mask + online softmax (rows m = r + 8*kh, key cols j = lane&15 [+16])
    const int j0 = jt + l15, j1 = jt + 16 + l15;
    float scf[8];
    #pragma unroll
    for (int r = 0; r < 8; ++r) {
      const int ig = q0 + r + 8 * kh;
      float x0 = (j0 <= ig && j0 > ig - WINW) ? s0[r] * scl : -1e30f;
      float x1 = (j1 <= ig && j1 > ig - WINW) ? s1[r] * scl : -1e30f;
      float tm = fmaxf(x0, x1);
      #pragma unroll
      for (int o = 1; o < 16; o <<= 1) tm = fmaxf(tm, __shfl_xor(tm, o, 32));
      const float mn = fmaxf(mrow[r], tm);
      const float sc = __expf(mrow[r] - mn);
      mrow[r] = mn;
      const float p0 = __expf(x0 - mn), p1 = __expf(x1 - mn);
      float ts = p0 + p1;
      #pragma unroll
      for (int o = 1; o < 16; o <<= 1) ts += __shfl_xor(ts, o, 32);
      lrow[r] = lrow[r] * sc + ts;
      scf[r] = sc;
      s0[r] = p0; s1[r] = p1;
    }
    #pragma unroll
    for (int t = 0; t < 8; ++t) {
      #pragma unroll
      for (int r = 0; r < 8; ++r) Oc[t][r] *= scf[r];
    }
    // C-layout P -> LDS -> A-layout P (per-wave private tile)
    #pragma unroll
    for (int r = 0; r < 8; ++r) {
      const int row = r + 8 * kh;
      Pl[row * 32 + l15]      = (__bf16)s0[r];
      Pl[row * 32 + 16 + l15] = (__bf16)s1[r];
    }
    asm volatile("s_wait_dscnt 0x0" ::: "memory");
    bfx8 plo = *(const bfx8*)(Pl + l15 * 32 + kh * 16);
    bfx8 phi = *(const bfx8*)(Pl + l15 * 32 + kh * 16 + 8);
    bfx16 pa = cat8(plo, phi);
    // ---- load all V operands, then 8 back-to-back WMMAs ----
    bfx16 vb[8];
    #pragma unroll
    for (int t = 0; t < 8; ++t)
      vb[t] = *(const bfx16*)(Vp + (size_t)(t * 16 + l15) * SS + jt + kh * 16);
    #pragma unroll
    for (int t = 0; t < 8; ++t)
      Oc[t] = wmma_bf16(pa, vb[t], Oc[t]);
    asm volatile("s_wait_dscnt 0x0" ::: "memory");
  }

  float inv[8];
  #pragma unroll
  for (int r = 0; r < 8; ++r) inv[r] = 1.0f / lrow[r];
  #pragma unroll
  for (int t = 0; t < 8; ++t) {
    #pragma unroll
    for (int r = 0; r < 8; ++r) {
      const int row = q0 + r + 8 * kh;
      Out[((size_t)b * SS + row) * (HH * DVV) + h * DVV + t * 16 + l15] =
          (__bf16)(Oc[t][r] * inv[r]);
    }
  }
}

// ---------------- host side ----------------
extern "C" void kernel_launch(void* const* d_in, const int* in_sizes, int n_in,
                              void* d_out, int out_size, void* d_ws, size_t ws_size,
                              hipStream_t stream) {
  (void)in_sizes; (void)n_in; (void)out_size; (void)ws_size;
  const float* emb  = (const float*)d_in[0];
  const float* cosb = (const float*)d_in[1];   // [2,1,S,DK]
  const float* sinb = (const float*)d_in[2];
  const float* wq   = (const float*)d_in[3];   // [E, H*DK]
  const float* wk   = (const float*)d_in[4];   // [E, HKV*DK]
  const float* wv   = (const float*)d_in[5];   // [E, HKV*DV]
  const float* wo   = (const float*)d_in[6];   // [H*DV, E]
  float* out = (float*)d_out;

  char* ws = (char*)d_ws;
  size_t off = 0;
  auto take = [&](size_t bytes) -> char* {
    char* p = ws + off;
    off = (off + bytes + 255) & ~(size_t)255;
    return p;
  };
  __bf16* normed = (__bf16*)take((size_t)MT * EE * 2);
  __bf16* wqT    = (__bf16*)take((size_t)EE * (HH  * DKK) * 2);
  __bf16* wkT    = (__bf16*)take((size_t)EE * (HKV * DKK) * 2);
  __bf16* wvT    = (__bf16*)take((size_t)EE * (HKV * DVV) * 2);
  __bf16* woT    = (__bf16*)take((size_t)(HH * DVV) * EE * 2);
  float*  qf     = (float*) take((size_t)MT * (HH  * DKK) * 4);
  float*  kf     = (float*) take((size_t)MT * (HKV * DKK) * 4);
  float*  vf     = (float*) take((size_t)MT * (HKV * DVV) * 4);
  __bf16* Qbf    = (__bf16*)take((size_t)BB * HH  * SS * DKK * 2);
  __bf16* Kbf    = (__bf16*)take((size_t)BB * HKV * SS * DKK * 2);
  __bf16* Vtbf   = (__bf16*)take((size_t)BB * HKV * DVV * SS * 2);
  __bf16* attnb  = (__bf16*)take((size_t)MT * (HH * DVV) * 2);

  // 1) RMSNorm
  rmsnorm_kernel<<<MT, 256, 0, stream>>>(emb, normed);

  // 2) weight cast+transpose
  {
    size_t n;
    n = (size_t)EE * (HH * DKK);
    cast_transpose_kernel<<<(n + 255) / 256, 256, 0, stream>>>(wq, wqT, EE, HH * DKK);
    n = (size_t)EE * (HKV * DKK);
    cast_transpose_kernel<<<(n + 255) / 256, 256, 0, stream>>>(wk, wkT, EE, HKV * DKK);
    n = (size_t)EE * (HKV * DVV);
    cast_transpose_kernel<<<(n + 255) / 256, 256, 0, stream>>>(wv, wvT, EE, HKV * DVV);
    n = (size_t)(HH * DVV) * EE;
    cast_transpose_kernel<<<(n + 255) / 256, 256, 0, stream>>>(wo, woT, HH * DVV, EE);
  }

  // 3) QKV projections (bf16 WMMA GEMM, fp32 out); N-tile is 128 wide
  gemm_bf16_wmma<<<dim3((HH * DKK) / 128, MT / 64), 128, 0, stream>>>(
      normed, wqT, qf, nullptr, HH * DKK, EE);
  gemm_bf16_wmma<<<dim3((HKV * DKK) / 128, MT / 64), 128, 0, stream>>>(
      normed, wkT, kf, nullptr, HKV * DKK, EE);
  gemm_bf16_wmma<<<dim3((HKV * DVV) / 128, MT / 64), 128, 0, stream>>>(
      normed, wvT, vf, nullptr, HKV * DVV, EE);

  // 4) RoPE + packs
  {
    size_t n;
    n = (size_t)BB * SS * HH * DKK;
    rope_pack_kernel<<<(n + 255) / 256, 256, 0, stream>>>(
        qf, cosb, sinb, Qbf, HH);                              // cos/sin[0]
    n = (size_t)BB * SS * HKV * DKK;
    rope_pack_kernel<<<(n + 255) / 256, 256, 0, stream>>>(
        kf, cosb + (size_t)SS * DKK, sinb + (size_t)SS * DKK, Kbf, HKV); // cos/sin[1]
    n = (size_t)BB * SS * HKV * DVV;
    pack_vT_kernel<<<(n + 255) / 256, 256, 0, stream>>>(vf, Vtbf);
  }

  // 5) sliding-window attention
  attn_kernel<<<dim3(SS / 64, HH, BB), 128, 0, stream>>>(Qbf, Kbf, Vtbf, attnb);

  // 6) output projection + residual
  gemm_bf16_wmma<<<dim3(EE / 128, MT / 64), 128, 0, stream>>>(
      attnb, woT, out, emb, EE, EE);
}
